// JitRBM_56289841381591
// MI455X (gfx1250) — matrix operand order repaired and checked by reading, not compiled, not run on hardware
//
#include <hip/hip_runtime.h>
#include <hip/hip_bf16.h>
#include <hip/hip_fp16.h>

// CDNA5 / gfx1250 conditional-RBM CD kernel.
// - All matmuls on v_wmma_f32_16x16x32_f16 (16 batch rows per wave, wave32)
// - v_data tiles DMA'd by the Tensor Data Mover (tensor_load_to_lds),
//   double-buffered across the Gibbs chain, synced with s_wait_tensorcnt
// - cond scalars staged with global_load_async_to_lds_b32 (ASYNCcnt)

typedef __attribute__((ext_vector_type(16))) _Float16 v16h;
typedef __attribute__((ext_vector_type(8)))  float    v8f;
typedef __attribute__((ext_vector_type(4)))  unsigned u32x4;
typedef __attribute__((ext_vector_type(8)))  unsigned u32x8;

#define NV      16
#define NH      32
#define HIDN    64
#define KSTEPS  5
#define NPARAM  96          // 2*(NV+NH)
#define NWAVES  4
#define THREADS (NWAVES * 32)
#define BTOT    1048576
#define NTILES  (BTOT / 16)
#define NBLOCKS 1024        // 4096 waves -> 16 tiles per wave

// ---------------------------------------------------------------- helpers --

__device__ __forceinline__ v8f wmma16(v16h a, v16h b, v8f c) {
  return __builtin_amdgcn_wmma_f32_16x16x32_f16(false, a, false, b,
                                                (short)0, c, false, false);
}

__device__ __forceinline__ void lds_fence() {
  asm volatile("s_wait_dscnt 0" ::: "memory");
}
__device__ __forceinline__ void wait_async0() {
  asm volatile("s_wait_asynccnt 0" ::: "memory");
}

__device__ __forceinline__ unsigned lds_off(const void* p) {
  // Generic LDS address: low 32 bits are the workgroup-relative LDS offset.
  return (unsigned)(unsigned long long)p;
}

// One ASYNCcnt-tracked dword: global -> LDS (per-lane, EXEC-predicated).
__device__ __forceinline__ void async_ld_b32(unsigned lds_byte, const float* g) {
  asm volatile("global_load_async_to_lds_b32 %0, %1, off"
               :: "v"(lds_byte), "v"(g) : "memory");
}

// TDM: DMA a 16x16 f32 row-major tile (stride NV) from v_data into LDS.
// 2-group D# (2D tensor): group0 = {count/flags, lds_addr, global_addr, type},
// group1 = dims/strides per CDNA5 ISA 8.3/8.4.
__device__ __forceinline__ void tdm_load_tile(const float* gsrc, const void* ldsdst) {
  unsigned long long ga = (unsigned long long)gsrc;
  u32x4 g0;
  g0[0] = 1u;                                           // count=1, no gather
  g0[1] = lds_off(ldsdst);                              // lds_addr
  g0[2] = (unsigned)ga;                                 // global_addr[31:0]
  g0[3] = (unsigned)((ga >> 32) & 0x01ffffffu) | (2u << 30); // addr[56:32] | type=2
  u32x8 g1;
  g1[0] = 0x00020000u;                                  // wg_mask=0, data_size=4B
  g1[1] = (unsigned)NV << 16;                           // tensor_dim0 = 16
  g1[2] = (unsigned)((BTOT & 0xffff) << 16);            // tensor_dim1 lo16 (=0)
  g1[3] = (unsigned)(BTOT >> 16) | (16u << 16);         // tensor_dim1 hi | tile_dim0=16
  g1[4] = 16u;                                          // tile_dim1=16, tile_dim2=0
  g1[5] = (unsigned)NV;                                 // tensor_dim0_stride = 16
  g1[6] = 0u;                                           // stride0 hi, stride1 lo
  g1[7] = 0u;                                           // stride1 hi
  asm volatile("tensor_load_to_lds %0, %1" :: "s"(g0), "s"(g1) : "memory");
}

// A-matrix (16x32, 16-bit) placement per CDNA5 ISA 7.12.2.
__device__ __forceinline__ int aSlot(int k)           { return (k & 7) | ((k >> 4) << 3); }
__device__ __forceinline__ int aLaneRow(int k, int m) { return (((k >> 3) & 1) << 4) + m; }
__device__ __forceinline__ int aInvK(int hi, int j)   { return (j & 7) + hi * 8 + ((j >> 3) << 4); }

__device__ __forceinline__ float rng01(unsigned r, unsigned u, unsigned s) {
  unsigned x = r * 0x9E3779B1u + u * 0x85EBCA77u + s * 0xC2B2AE3Du + 0x27220A95u;
  x ^= x >> 16; x *= 0x7FEB352Du;
  x ^= x >> 15; x *= 0x846CA68Bu;
  x ^= x >> 16;
  return (float)(x >> 8) * (1.0f / 16777216.0f);
}

__device__ __forceinline__ float fast_tanh(float x) {
#if __has_builtin(__builtin_amdgcn_tanhf)
  return __builtin_amdgcn_tanhf(x);           // v_tanh_f32 on gfx1250
#else
  return tanhf(x);
#endif
}
__device__ __forceinline__ float sigmoidf(float x) {
  return __frcp_rn(1.0f + __expf(-x));
}
__device__ __forceinline__ float softplusf(float x) {
  return fmaxf(x, 0.0f) + __logf(1.0f + __expf(-fabsf(x)));
}

// ---------------------------------------------------------------- kernels --

__global__ void rbm_zero_out(float* out) { out[0] = 0.0f; }

__global__ __launch_bounds__(THREADS)
void rbm_cd_kernel(const float* __restrict__ v_data, const float* __restrict__ cond,
                   const float* __restrict__ W,      const float* __restrict__ b,
                   const float* __restrict__ c,      const float* __restrict__ w1,
                   const float* __restrict__ b1,     const float* __restrict__ w2,
                   const float* __restrict__ b2,     float* __restrict__ out) {
  __shared__ v16h sW_B[2][32];            // V@W : two 16-col tiles, K pad 16->32
  __shared__ v16h sWt_B[32];              // H@W^T : K=32, N=16
  __shared__ v16h sw2_B[12][32];          // cond-net layer2 operands
  __shared__ float s_w1[HIDN], s_b1[HIDN], s_b2[NPARAM], s_b[NV], s_c[NH];
  __shared__ v16h  sV[NWAVES][32];        // current v as A-matrix (16x32 f16)
  __shared__ v16h  sH[NWAVES][32];        // sampled h as A-matrix
  __shared__ float s_bmod[NWAVES][16][NV];
  __shared__ float s_cmod[NWAVES][16][NH];
  __shared__ float s_feD[NWAVES][16];
  __shared__ float s_feM[NWAVES][16];
  __shared__ __align__(16) float s_vraw[NWAVES][2][16][NV];  // TDM double buffer
  __shared__ float s_craw[NWAVES][2][16];                    // async cond buffer

  const int tid  = threadIdx.x;
  const int lane = tid & 31;
  const int wv   = tid >> 5;
  const int hi   = lane >> 4;
  const int nn   = lane & 15;
  const int m8   = hi * 8;

  // ---- one-time weight staging ------------------------------------------
  {
    _Float16* hW = (_Float16*)&sW_B[0][0];
    for (int idx = tid; idx < 2 * 32 * 16; idx += THREADS) {
      int t = idx >> 9, ln = (idx >> 4) & 31, j = idx & 15;
      int k = (j & 7) + ((ln >> 4) << 3) + ((j >> 3) << 4);
      int n = t * 16 + (ln & 15);
      hW[idx] = (k < NV) ? (_Float16)W[k * NH + n] : (_Float16)0.0f;
    }
    _Float16* hWt = (_Float16*)&sWt_B[0];
    for (int idx = tid; idx < 32 * 16; idx += THREADS) {
      int ln = (idx >> 4) & 31, j = idx & 15;
      int k = (j & 7) + ((ln >> 4) << 3) + ((j >> 3) << 4);
      int n = ln & 15;
      hWt[idx] = (_Float16)W[n * NH + k];
    }
    _Float16* hw2 = (_Float16*)&sw2_B[0][0];
    for (int idx = tid; idx < 12 * 32 * 16; idx += THREADS) {
      int op = idx >> 9, ln = (idx >> 4) & 31, j = idx & 15;
      int nt = op >> 1, kh = op & 1;
      int k = kh * 32 + (j & 7) + ((ln >> 4) << 3) + ((j >> 3) << 4);
      int n = nt * 16 + (ln & 15);
      hw2[idx] = (_Float16)w2[k * NPARAM + n];
    }
    for (int i = tid; i < HIDN;   i += THREADS) { s_w1[i] = w1[i]; s_b1[i] = b1[i]; }
    for (int i = tid; i < NPARAM; i += THREADS) { s_b2[i] = b2[i]; }
    for (int i = tid; i < NV;     i += THREADS) { s_b[i] = b[i]; }
    for (int i = tid; i < NH;     i += THREADS) { s_c[i] = c[i]; }
  }
  __syncthreads();

  // Loop-invariant B operands live in VGPRs for the whole chain.
  const v16h wB0 = sW_B[0][lane];
  const v16h wB1 = sW_B[1][lane];
  const v16h wtB = sWt_B[lane];

  const int wave_id  = blockIdx.x * NWAVES + wv;
  const int wave_cnt = gridDim.x * NWAVES;

  // ---- prologue: kick DMA for first tile --------------------------------
  int buf = 0;
  if (wave_id < NTILES) {
    tdm_load_tile(v_data + (size_t)wave_id * 16 * NV, &s_vraw[wv][0][0][0]);
    if (lane < 16)
      async_ld_b32(lds_off(&s_craw[wv][0][lane]), cond + wave_id * 16 + lane);
  }

  for (int tile = wave_id; tile < NTILES; tile += wave_cnt) {
    const int row0 = tile * 16;
    __builtin_amdgcn_s_wait_tensorcnt(0);   // current v_data tile in LDS
    wait_async0();                          // current cond scalars in LDS

    // ---- pack v_data into A-matrix layout (zero-pad K 16..31) ------------
    {
      _Float16* dst = (_Float16*)&sV[wv][lane];
      #pragma unroll
      for (int j = 0; j < 16; ++j) {
        int k = aInvK(hi, j);
        dst[j] = (k < NV) ? (_Float16)s_vraw[wv][buf][nn][k] : (_Float16)0.0f;
      }
    }

    // ---- cond-net layer1 (v_tanh) straight into A operands ---------------
    float cm = s_craw[wv][buf][nn];
    v16h a0, a1;
    #pragma unroll
    for (int j = 0; j < 16; ++j) {
      int k0 = aInvK(hi, j);
      a0[j] = (_Float16)fast_tanh(fmaf(cm, s_w1[k0],      s_b1[k0]));
      a1[j] = (_Float16)fast_tanh(fmaf(cm, s_w1[32 + k0], s_b1[32 + k0]));
    }
    // layer2: 6 output tiles x (K=64 -> 2 WMMAs)
    v8f P[6];
    #pragma unroll
    for (int nt = 0; nt < 6; ++nt) {
      v8f acc = {};
      acc = wmma16(a0, sw2_B[nt * 2 + 0][lane], acc);
      acc = wmma16(a1, sw2_B[nt * 2 + 1][lane], acc);
      P[nt] = acc;
    }

    // ---- FiLM modulation --------------------------------------------------
    #pragma unroll
    for (int r = 0; r < 8; ++r) {
      int m = m8 + r;
      float gb  = P[0][r] + s_b2[nn];
      float bb  = P[1][r] + s_b2[16 + nn];
      s_bmod[wv][m][nn] = (1.0f + gb) * s_b[nn] + bb;
      float gc0 = P[2][r] + s_b2[32 + nn];
      float bc0 = P[4][r] + s_b2[64 + nn];
      s_cmod[wv][m][nn] = (1.0f + gc0) * s_c[nn] + bc0;
      float gc1 = P[3][r] + s_b2[48 + nn];
      float bc1 = P[5][r] + s_b2[80 + nn];
      s_cmod[wv][m][16 + nn] = (1.0f + gc1) * s_c[16 + nn] + bc1;
    }
    if (lane < 16) { s_feD[wv][lane] = 0.0f; s_feM[wv][lane] = 0.0f; }
    lds_fence();

    // ---- free energy of data ---------------------------------------------
    {
      v16h av = sV[wv][lane];
      v8f X0 = {}, X1 = {};
      X0 = wmma16(av, wB0, X0);
      X1 = wmma16(av, wB1, X1);
      #pragma unroll
      for (int r = 0; r < 8; ++r) {
        int m = m8 + r;
        float acc = -softplusf(X0[r] + s_cmod[wv][m][nn])
                    -softplusf(X1[r] + s_cmod[wv][m][16 + nn])
                    -s_vraw[wv][buf][m][nn] * s_bmod[wv][m][nn];
        atomicAdd(&s_feD[wv][m], acc);
      }
    }

    // ---- last consumer of staging buffers done: DMA the next tile --------
    {
      int nxt = tile + wave_cnt;
      lds_fence();     // TDM is unordered vs DS: drain reads of s_vraw/s_craw
      if (nxt < NTILES) {
        tdm_load_tile(v_data + (size_t)nxt * 16 * NV, &s_vraw[wv][buf ^ 1][0][0]);
        if (lane < 16)
          async_ld_b32(lds_off(&s_craw[wv][buf ^ 1][lane]), cond + nxt * 16 + lane);
        if (nxt + wave_cnt < NTILES)
          __builtin_prefetch(v_data + (size_t)(nxt + wave_cnt) * 16 * NV, 0, 1);
      }
    }

    // ---- CD-k Gibbs chain (WMMA B operands stay in VGPRs) ----------------
    for (int step = 0; step < KSTEPS; ++step) {
      v16h av = sV[wv][lane];
      v8f H0 = {}, H1 = {};
      H0 = wmma16(av, wB0, H0);
      H1 = wmma16(av, wB1, H1);
      _Float16* hdst = (_Float16*)&sH[wv][0];
      #pragma unroll
      for (int r = 0; r < 8; ++r) {
        int m = m8 + r;
        float p0 = sigmoidf(H0[r] + s_cmod[wv][m][nn]);
        float h0 = (rng01(row0 + m, nn, 2 * step) < p0) ? 1.0f : 0.0f;
        hdst[aLaneRow(nn, m) * 16 + aSlot(nn)] = (_Float16)h0;
        float p1 = sigmoidf(H1[r] + s_cmod[wv][m][16 + nn]);
        float h1 = (rng01(row0 + m, 16 + nn, 2 * step) < p1) ? 1.0f : 0.0f;
        hdst[aLaneRow(16 + nn, m) * 16 + aSlot(16 + nn)] = (_Float16)h1;
      }
      lds_fence();

      v16h ah = sH[wv][lane];
      v8f V0 = {};
      V0 = wmma16(ah, wtB, V0);
      _Float16* vdst = (_Float16*)&sV[wv][0];
      #pragma unroll
      for (int r = 0; r < 8; ++r) {
        int m = m8 + r;
        float p  = sigmoidf(V0[r] + s_bmod[wv][m][nn]);
        float vb = (rng01(row0 + m, nn, 2 * step + 1) < p) ? 1.0f : 0.0f;
        vdst[aLaneRow(nn, m) * 16 + aSlot(nn)] = (_Float16)vb;  // K>=16 stays 0
      }
      lds_fence();
    }

    // ---- free energy of model sample -------------------------------------
    {
      v16h av = sV[wv][lane];
      v8f X0 = {}, X1 = {};
      X0 = wmma16(av, wB0, X0);
      X1 = wmma16(av, wB1, X1);
      const _Float16* vh = (const _Float16*)&sV[wv][0];
      #pragma unroll
      for (int r = 0; r < 8; ++r) {
        int m = m8 + r;
        float vmv = (float)vh[aLaneRow(nn, m) * 16 + aSlot(nn)];
        float acc = -softplusf(X0[r] + s_cmod[wv][m][nn])
                    -softplusf(X1[r] + s_cmod[wv][m][16 + nn])
                    -vmv * s_bmod[wv][m][nn];
        atomicAdd(&s_feM[wv][m], acc);
      }
    }
    lds_fence();

    // ---- per-tile contribution to mean(fe_data - fe_model) ---------------
    float diff = (lane < 16) ? (s_feD[wv][lane] - s_feM[wv][lane]) : 0.0f;
    #pragma unroll
    for (int o = 16; o >= 1; o >>= 1) diff += __shfl_xor(diff, o, 32);
    if (lane == 0) atomicAdd(out, diff * (1.0f / (float)BTOT));

    buf ^= 1;
  }
}

// ---------------------------------------------------------------- launch ---

extern "C" void kernel_launch(void* const* d_in, const int* in_sizes, int n_in,
                              void* d_out, int out_size, void* d_ws, size_t ws_size,
                              hipStream_t stream) {
  (void)in_sizes; (void)n_in; (void)d_ws; (void)ws_size; (void)out_size;
  const float* v_data = (const float*)d_in[0];
  const float* cond   = (const float*)d_in[1];
  const float* W      = (const float*)d_in[2];
  const float* b      = (const float*)d_in[3];
  const float* c      = (const float*)d_in[4];
  const float* w1     = (const float*)d_in[5];
  const float* b1     = (const float*)d_in[6];
  const float* w2     = (const float*)d_in[7];
  const float* b2     = (const float*)d_in[8];
  float* out = (float*)d_out;

  rbm_zero_out<<<1, 1, 0, stream>>>(out);
  rbm_cd_kernel<<<NBLOCKS, THREADS, 0, stream>>>(v_data, cond, W, b, c,
                                                 w1, b1, w2, b2, out);
}
